// DenoiseGRU_29497835389343
// MI455X (gfx1250) — matrix operand order repaired and checked by reading, not compiled, not run on hardware
//
#include <hip/hip_runtime.h>

#define DEV_INLINE __device__ __forceinline__

typedef __attribute__((ext_vector_type(16))) _Float16 v16h;
typedef __attribute__((ext_vector_type(8)))  _Float16 v8h;
typedef __attribute__((ext_vector_type(2)))  _Float16 v2h;
typedef __attribute__((ext_vector_type(8)))  float    v8f;
typedef __attribute__((ext_vector_type(4)))  int      v4i;

static constexpr int B = 32, S = 128, T = 128, V = 32000;
static constexpr int DE = 128, DENC = 256, DDEC = 512, HD = 64;

#if __has_builtin(__builtin_amdgcn_global_load_async_to_lds_b128) && \
    __has_builtin(__builtin_amdgcn_s_wait_asynccnt)
#define ASYNC_LDS 1
typedef __attribute__((address_space(1))) v4i as1_v4i;
typedef __attribute__((address_space(3))) v4i as3_v4i;
#else
#define ASYNC_LDS 0
#endif

DEV_INLINE float sigmoidf_(float x) { return 1.f / (1.f + __expf(-x)); }

// ---- WMMA fragment loaders (CDNA5 ISA 7.12.2, wave32, 16-bit A/B, f32 C/D) ----
DEV_INLINE v16h load_frag_a(const _Float16* rowbase, int hsel) {
  v16h a;
#pragma unroll
  for (int g = 0; g < 8; ++g) {
    const int ka = (g < 4 ? 2 * g : 16 + 2 * (g - 4)) + (hsel ? 8 : 0);
    v2h p = *(const v2h*)(rowbase + ka);
    a[2 * g] = p[0]; a[2 * g + 1] = p[1];
  }
  return a;
}
DEV_INLINE v16h load_frag_b(const _Float16* rowbase, int hsel) {
  v16h b;
#pragma unroll
  for (int g = 0; g < 8; ++g) {
    const int kb = (g < 4 ? 2 * g : 16 + 2 * (g - 4)) + (hsel ? 8 : 0);
    v2h p = *(const v2h*)(rowbase + kb);
    b[2 * g] = p[0]; b[2 * g + 1] = p[1];
  }
  return b;
}

// One wave computes a 16x16 f32 tile of  A[MxK] @ W[NxK]^T  accumulating into c.
DEV_INLINE v8f wmma_tile(const _Float16* A, int lda, const _Float16* W, int ldw,
                         int m0, int n0, int K, v8f c) {
  const int lane = threadIdx.x & 31;
  const int hsel = lane >> 4;
  const int r    = lane & 15;
  for (int k0 = 0; k0 < K; k0 += 32) {
    v16h a = load_frag_a(A + (size_t)(m0 + r) * lda + k0, hsel);
    v16h b = load_frag_b(W + (size_t)(n0 + r) * ldw + k0, hsel);
    c = __builtin_amdgcn_wmma_f32_16x16x32_f16(false, a, false, b, (short)0, c,
                                               false, false);
  }
  return c;
}

// ---------------- small utility kernels ----------------
__global__ void k_f32_to_f16(const float* __restrict__ s, _Float16* __restrict__ d, int n) {
  for (int i = blockIdx.x * blockDim.x + threadIdx.x; i < n; i += gridDim.x * blockDim.x)
    d[i] = (_Float16)s[i];
}

__global__ void k_gather_enc(const int* __restrict__ nx, const float* __restrict__ tok,
                             _Float16* __restrict__ o) {
  const int n = S * B * DE;
  for (int i = blockIdx.x * blockDim.x + threadIdx.x; i < n; i += gridDim.x * blockDim.x) {
    const int d = i & (DE - 1), m = i >> 7;
    const int b = m & (B - 1), s = m >> 5;
    o[i] = (_Float16)tok[(size_t)nx[b * S + s] * DE + d];
  }
}

__global__ void k_gather_dec(const int* __restrict__ x, const float* __restrict__ tok,
                             const float* __restrict__ start, _Float16* __restrict__ o) {
  const int n = T * B * DE;
  for (int i = blockIdx.x * blockDim.x + threadIdx.x; i < n; i += gridDim.x * blockDim.x) {
    const int d = i & (DE - 1), m = i >> 7;
    const int b = m & (B - 1), t = m >> 5;
    const float v = (t == 0) ? start[d] : tok[(size_t)x[b * T + t - 1] * DE + d];
    o[i] = (_Float16)v;
  }
}

// ---------------- generic WMMA GEMM:  out(f16) = A @ W^T + bias ----------------
__global__ __launch_bounds__(256) void k_gemm_f16(
    const _Float16* __restrict__ A, int lda, const _Float16* __restrict__ W, int ldw,
    const float* __restrict__ bias, _Float16* __restrict__ out, int ldo,
    int M, int N, int K) {
  const int lane = threadIdx.x & 31, hsel = lane >> 4, r = lane & 15;
  const int wv = blockIdx.x * (blockDim.x >> 5) + (threadIdx.x >> 5);
  const int nw = gridDim.x * (blockDim.x >> 5);
  const int tilesN = N >> 4, total = (M >> 4) * tilesN;
  for (int tile = wv; tile < total; tile += nw) {
    const int m0 = (tile / tilesN) << 4, n0 = (tile % tilesN) << 4;
    v8f c;
    const float bv = bias ? bias[n0 + r] : 0.f;
#pragma unroll
    for (int i = 0; i < 8; ++i) c[i] = bv;
    c = wmma_tile(A, lda, W, ldw, m0, n0, K, c);
#pragma unroll
    for (int i = 0; i < 8; ++i)
      out[(size_t)(m0 + i + (hsel ? 8 : 0)) * ldo + n0 + r] = (_Float16)c[i];
  }
}

// ------- final projection: logits[B,T,V] = hs @ proj_w^T + b -------
// LDS-tiled macro-block GEMM: block = 128x128 output, K chunks of 32 double-
// buffered in LDS (async global->LDS when the toolchain exposes the builtin).
// 8 waves, each computing a 2(M) x 4(N) grid of 16x16 WMMA tiles so every
// A fragment is reused 4x and every B fragment 2x from LDS.
__global__ __launch_bounds__(256) void k_gemm_proj(
    const _Float16* __restrict__ A, const _Float16* __restrict__ W,
    const float* __restrict__ bias, float* __restrict__ out) {
  __shared__ _Float16 As[2][128 * 32];   // 8 KB x2
  __shared__ _Float16 Bs[2][128 * 32];   // 8 KB x2
  const int tid  = threadIdx.x;
  const int lane = tid & 31, hsel = lane >> 4, r = lane & 15;
  const int wv   = tid >> 5;
  const int mi   = wv >> 1;              // 0..3: M subtiles 2*mi, 2*mi+1
  const int ni   = wv & 1;               // 0..1: N subtiles 4*ni .. 4*ni+3
  const int bm   = blockIdx.x & 31;      // 32 M blocks (M = 4096)
  const int bn   = blockIdx.x >> 5;      // 250 N blocks (N = 32000)
  const int M0   = bm << 7, N0 = bn << 7;
  const int srow = tid >> 1;             // staging: 128 rows x 32 halfs
  const int scol = (tid & 1) << 4;       // two 16-half (32B) chunks per row

  auto stage = [&](int buf, int k0) {
    const _Float16* gA = A + (size_t)(M0 + srow) * DDEC + k0 + scol;
    const _Float16* gW = W + (size_t)(N0 + srow) * DDEC + k0 + scol;
    const int l = srow * 32 + scol;
#if ASYNC_LDS
    __builtin_amdgcn_global_load_async_to_lds_b128(
        (as1_v4i*)gA, (as3_v4i*)&As[buf][l], 0, 0);
    __builtin_amdgcn_global_load_async_to_lds_b128(
        (as1_v4i*)gW, (as3_v4i*)&Bs[buf][l], 0, 0);
#else
    *(v8h*)&As[buf][l] = *(const v8h*)gA;
    *(v8h*)&Bs[buf][l] = *(const v8h*)gW;
#endif
  };

  v8f acc[8];
#pragma unroll
  for (int tn = 0; tn < 4; ++tn) {
    const float bv = bias[N0 + ((ni * 4 + tn) << 4) + r];
#pragma unroll
    for (int i = 0; i < 8; ++i) { acc[tn][i] = bv; acc[4 + tn][i] = bv; }
  }

  stage(0, 0);
  for (int kc = 0; kc < 16; ++kc) {
    const int buf = kc & 1;
#if ASYNC_LDS
    __builtin_amdgcn_s_wait_asynccnt(0);
#endif
    __syncthreads();
    if (kc < 15) stage(buf ^ 1, (kc + 1) * 32);
    v16h af0 = load_frag_a(&As[buf][(((mi * 2 + 0) << 4) + r) * 32], hsel);
    v16h af1 = load_frag_a(&As[buf][(((mi * 2 + 1) << 4) + r) * 32], hsel);
#pragma unroll
    for (int tn = 0; tn < 4; ++tn) {
      v16h bf = load_frag_b(&Bs[buf][(((ni * 4 + tn) << 4) + r) * 32], hsel);
      acc[tn]     = __builtin_amdgcn_wmma_f32_16x16x32_f16(false, af0, false, bf,
                                                           (short)0, acc[tn], false, false);
      acc[4 + tn] = __builtin_amdgcn_wmma_f32_16x16x32_f16(false, af1, false, bf,
                                                           (short)0, acc[4 + tn], false, false);
    }
    __syncthreads();
  }
#pragma unroll
  for (int tm = 0; tm < 2; ++tm)
#pragma unroll
    for (int tn = 0; tn < 4; ++tn)
#pragma unroll
      for (int i = 0; i < 8; ++i) {
        const int m = M0 + ((mi * 2 + tm) << 4) + i + (hsel ? 8 : 0);
        const int n = N0 + ((ni * 4 + tn) << 4) + r;
        const int tt = m >> 5, bb = m & 31;
        out[(size_t)bb * T * V + (size_t)tt * V + n] = acc[tm * 4 + tn][i];
      }
}

// ---------------- persistent bidirectional encoder GRU scan ----------------
__global__ __launch_bounds__(256) void k_enc_scan(
    const _Float16* __restrict__ Whh_f, const _Float16* __restrict__ Whh_b,
    const float* __restrict__ bhh_f, const float* __restrict__ bhh_b,
    const _Float16* __restrict__ gi_f, const _Float16* __restrict__ gi_b,
    _Float16* __restrict__ mem16) {
  __shared__ float    h32[B * DENC];        // 32 KB
  __shared__ _Float16 h16[B * DENC];        // 16 KB
  __shared__ float    gh[B * 3 * DENC];     // 96 KB
  const int dir = blockIdx.x;
  const _Float16* Whh = dir ? Whh_b : Whh_f;
  const float*    bhh = dir ? bhh_b : bhh_f;
  const _Float16* gi  = dir ? gi_b  : gi_f;
  const int tid = threadIdx.x;
  const int wv = tid >> 5, lane = tid & 31, hsel = lane >> 4, r = lane & 15;
  for (int i = tid; i < B * DENC; i += 256) h32[i] = 0.f;
  __syncthreads();
  for (int step = 0; step < S; ++step) {
    const int s = dir ? (S - 1 - step) : step;
    for (int i = tid; i < B * DENC; i += 256) h16[i] = (_Float16)h32[i];
    __syncthreads();
    for (int tile = wv; tile < 2 * 48; tile += 8) {
      const int m0 = (tile / 48) << 4, n0 = (tile % 48) << 4;
      v8f c;
      const float bv = bhh[n0 + r];
#pragma unroll
      for (int i = 0; i < 8; ++i) c[i] = bv;
      c = wmma_tile(h16, DENC, Whh, DENC, m0, n0, DENC, c);
#pragma unroll
      for (int i = 0; i < 8; ++i)
        gh[(m0 + i + (hsel ? 8 : 0)) * (3 * DENC) + n0 + r] = c[i];
    }
    __syncthreads();
    for (int i = tid; i < B * DENC; i += 256) {
      const int bb = i >> 8, j = i & 255;
      const _Float16* g = gi + (size_t)(s * B + bb) * (3 * DENC);
      const float ir = (float)g[j], iz = (float)g[DENC + j], inn = (float)g[2 * DENC + j];
      const float* gr = gh + bb * (3 * DENC);
      const float hr = gr[j], hz = gr[DENC + j], hn = gr[2 * DENC + j];
      const float rg = sigmoidf_(ir + hr);
      const float zg = sigmoidf_(iz + hz);
      const float ng = tanhf(inn + rg * hn);
      const float hnew = (1.f - zg) * ng + zg * h32[i];
      h32[i] = hnew;
      mem16[(size_t)(s * B + bb) * DDEC + dir * DENC + j] = (_Float16)hnew;
    }
    __syncthreads();
  }
}

// ---------------- persistent decoder scan: GRU + attention + LN ----------------
__global__ __launch_bounds__(512) void k_dec_scan(
    const _Float16* __restrict__ Whh, const float* __restrict__ bhh,
    const _Float16* __restrict__ gi,
    const _Float16* __restrict__ Wq, const float* __restrict__ bq,
    const _Float16* __restrict__ Wo, const float* __restrict__ bo,
    const _Float16* __restrict__ kbuf, const _Float16* __restrict__ vbuf,
    const float* __restrict__ lng, const float* __restrict__ lnb,
    const float* __restrict__ style, const int* __restrict__ label,
    float* __restrict__ ghbuf, _Float16* __restrict__ hs16) {
  __shared__ float    h32[B * DDEC];        // 64 KB
  __shared__ _Float16 xh16[B * DDEC];       // 32 KB
  __shared__ _Float16 qc16[B * DDEC];       // 32 KB
  const int tid = threadIdx.x;
  const int wv = tid >> 5, lane = tid & 31, hsel = lane >> 4, r = lane & 15;
  for (int i = tid; i < B * DDEC; i += 512) {
    const int bb = i >> 9, d = i & 511;
    h32[i] = style[(size_t)label[bb] * DDEC + d];
  }
  __syncthreads();
  for (int t = 0; t < T; ++t) {
    for (int i = tid; i < B * DDEC; i += 512) xh16[i] = (_Float16)h32[i];
    __syncthreads();
    for (int tile = wv; tile < 2 * 96; tile += 16) {
      const int m0 = (tile / 96) << 4, n0 = (tile % 96) << 4;
      v8f c;
      const float bv = bhh[n0 + r];
#pragma unroll
      for (int i = 0; i < 8; ++i) c[i] = bv;
      c = wmma_tile(xh16, DDEC, Whh, DDEC, m0, n0, DDEC, c);
#pragma unroll
      for (int i = 0; i < 8; ++i)
        ghbuf[(m0 + i + (hsel ? 8 : 0)) * (3 * DDEC) + n0 + r] = c[i];
    }
    __threadfence();
    __syncthreads();
    for (int i = tid; i < B * DDEC; i += 512) {
      const int bb = i >> 9, j = i & 511;
      const _Float16* g = gi + (size_t)(t * B + bb) * (3 * DDEC);
      const float ir = (float)g[j], iz = (float)g[DDEC + j], inn = (float)g[2 * DDEC + j];
      const float* gr = ghbuf + bb * (3 * DDEC);
      const float hr = gr[j], hz = gr[DDEC + j], hn = gr[2 * DDEC + j];
      const float rg = sigmoidf_(ir + hr);
      const float zg = sigmoidf_(iz + hz);
      const float ng = tanhf(inn + rg * hn);
      const float hd = (1.f - zg) * ng + zg * h32[i];
      h32[i] = hd;
      xh16[i] = (_Float16)hd;
    }
    __syncthreads();
    for (int tile = wv; tile < 2 * 32; tile += 16) {
      const int m0 = (tile / 32) << 4, n0 = (tile % 32) << 4;
      v8f c;
      const float bv = bq[n0 + r];
#pragma unroll
      for (int i = 0; i < 8; ++i) c[i] = bv;
      c = wmma_tile(xh16, DDEC, Wq, DDEC, m0, n0, DDEC, c);
#pragma unroll
      for (int i = 0; i < 8; ++i)
        qc16[(m0 + i + (hsel ? 8 : 0)) * DDEC + n0 + r] = (_Float16)c[i];
    }
    __syncthreads();
    for (int pp = 0; pp < 16; ++pp) {
      const int pair = wv * 16 + pp;
      const int bb = pair >> 3, hh = pair & 7;
      const _Float16* qrow = qc16 + bb * DDEC + hh * HD;
      const float q0 = (float)qrow[2 * lane];
      const float q1 = (float)qrow[2 * lane + 1];
      float sc[4] = {0.f, 0.f, 0.f, 0.f};
      const _Float16* kb = kbuf + (size_t)bb * DDEC + hh * HD;
      for (int l2 = 0; l2 < 32; ++l2) {
        const float qa = __shfl(q0, l2, 32);
        const float qb = __shfl(q1, l2, 32);
        const int d0 = 2 * l2;
#pragma unroll
        for (int i2 = 0; i2 < 4; ++i2) {
          const _Float16* kp = kb + (size_t)(lane + 32 * i2) * (B * DDEC) + d0;
          sc[i2] += qa * (float)kp[0] + qb * (float)kp[1];
        }
      }
#pragma unroll
      for (int i2 = 0; i2 < 4; ++i2) sc[i2] *= 0.125f;
      float mx = fmaxf(fmaxf(sc[0], sc[1]), fmaxf(sc[2], sc[3]));
      for (int o = 16; o; o >>= 1) mx = fmaxf(mx, __shfl_xor(mx, o, 32));
      float ssum = 0.f;
#pragma unroll
      for (int i2 = 0; i2 < 4; ++i2) { sc[i2] = __expf(sc[i2] - mx); ssum += sc[i2]; }
      for (int o = 16; o; o >>= 1) ssum += __shfl_xor(ssum, o, 32);
      const float inv = 1.f / ssum;
#pragma unroll
      for (int i2 = 0; i2 < 4; ++i2) sc[i2] *= inv;
      const int d0 = 2 * lane;
      const _Float16* vb = vbuf + (size_t)bb * DDEC + hh * HD + d0;
      float a0 = 0.f, a1 = 0.f;
#pragma unroll
      for (int i2 = 0; i2 < 4; ++i2)
        for (int l2 = 0; l2 < 32; ++l2) {
          const float aw = __shfl(sc[i2], l2, 32);
          const _Float16* vp = vb + (size_t)(l2 + 32 * i2) * (B * DDEC);
          a0 += aw * (float)vp[0];
          a1 += aw * (float)vp[1];
        }
      _Float16* crow = qc16 + bb * DDEC + hh * HD;
      crow[d0] = (_Float16)a0;
      crow[d0 + 1] = (_Float16)a1;
    }
    __syncthreads();
    for (int tile = wv; tile < 2 * 32; tile += 16) {
      const int m0 = (tile / 32) << 4, n0 = (tile % 32) << 4;
      v8f c;
#pragma unroll
      for (int i = 0; i < 8; ++i) {
        const int m = m0 + i + (hsel ? 8 : 0);
        c[i] = h32[m * DDEC + n0 + r] + bo[n0 + r];
      }
      c = wmma_tile(qc16, DDEC, Wo, DDEC, m0, n0, DDEC, c);
#pragma unroll
      for (int i = 0; i < 8; ++i)
        h32[(m0 + i + (hsel ? 8 : 0)) * DDEC + n0 + r] = c[i];
    }
    __syncthreads();
    {
      const int row = wv * 2 + hsel;
      float yv[32], lsum = 0.f, lsq = 0.f;
#pragma unroll
      for (int j = 0; j < 32; ++j) {
        const float y = h32[row * DDEC + r + 16 * j];
        yv[j] = y; lsum += y; lsq += y * y;
      }
      for (int o = 8; o; o >>= 1) {
        lsum += __shfl_xor(lsum, o, 16);
        lsq  += __shfl_xor(lsq,  o, 16);
      }
      const float mu  = lsum * (1.f / 512.f);
      const float var = lsq * (1.f / 512.f) - mu * mu;
      const float rin = rsqrtf(var + 1e-5f);
#pragma unroll
      for (int j = 0; j < 32; ++j) {
        const int d = r + 16 * j;
        const float hn2 = (yv[j] - mu) * rin * lng[d] + lnb[d];
        h32[row * DDEC + d] = hn2;
        hs16[(size_t)(t * B + row) * DDEC + d] = (_Float16)hn2;
      }
    }
    __syncthreads();
  }
}

extern "C" void kernel_launch(void* const* d_in, const int* in_sizes, int n_in,
                              void* d_out, int out_size, void* d_ws, size_t ws_size,
                              hipStream_t stream) {
  (void)in_sizes; (void)n_in; (void)out_size; (void)ws_size;
  const int*   nx        = (const int*)d_in[0];
  const int*   x         = (const int*)d_in[1];
  const int*   label     = (const int*)d_in[2];
  const float* tok_emb   = (const float*)d_in[3];
  const float* start_emb = (const float*)d_in[4];
  const float* style_emb = (const float*)d_in[5];
  const float* enc_Wih_f = (const float*)d_in[6];
  const float* enc_Whh_f = (const float*)d_in[7];
  const float* enc_bih_f = (const float*)d_in[8];
  const float* enc_bhh_f = (const float*)d_in[9];
  const float* enc_Wih_b = (const float*)d_in[10];
  const float* enc_Whh_b = (const float*)d_in[11];
  const float* enc_bih_b = (const float*)d_in[12];
  const float* enc_bhh_b = (const float*)d_in[13];
  const float* dec_Wih   = (const float*)d_in[14];
  const float* dec_Whh   = (const float*)d_in[15];
  const float* dec_bih   = (const float*)d_in[16];
  const float* dec_bhh   = (const float*)d_in[17];
  const float* attn_in_w = (const float*)d_in[18];
  const float* attn_in_b = (const float*)d_in[19];
  const float* attn_out_w= (const float*)d_in[20];
  const float* attn_out_b= (const float*)d_in[21];
  const float* ln_g      = (const float*)d_in[22];
  const float* ln_b      = (const float*)d_in[23];
  const float* proj_w    = (const float*)d_in[24];
  const float* proj_b    = (const float*)d_in[25];
  float* out = (float*)d_out;

  char* base = (char*)d_ws;
  size_t off = 0;
  auto alloc = [&](size_t bytes) -> void* {
    void* p = base + off;
    off = (off + bytes + 255) & ~(size_t)255;
    return p;
  };
  _Float16* encWihF16 = (_Float16*)alloc((size_t)768 * 128 * 2);
  _Float16* encWhhF16 = (_Float16*)alloc((size_t)768 * 256 * 2);
  _Float16* encWihB16 = (_Float16*)alloc((size_t)768 * 128 * 2);
  _Float16* encWhhB16 = (_Float16*)alloc((size_t)768 * 256 * 2);
  _Float16* decWih16  = (_Float16*)alloc((size_t)1536 * 128 * 2);
  _Float16* decWhh16  = (_Float16*)alloc((size_t)1536 * 512 * 2);
  _Float16* attnIn16  = (_Float16*)alloc((size_t)1536 * 512 * 2);
  _Float16* attnOut16 = (_Float16*)alloc((size_t)512 * 512 * 2);
  _Float16* projW16   = (_Float16*)alloc((size_t)32000 * 512 * 2);
  _Float16* emb16     = (_Float16*)alloc((size_t)S * B * DE * 2);
  _Float16* decin16   = (_Float16*)alloc((size_t)T * B * DE * 2);
  _Float16* giF16     = (_Float16*)alloc((size_t)S * B * 3 * DENC * 2);
  _Float16* giB16     = (_Float16*)alloc((size_t)S * B * 3 * DENC * 2);
  _Float16* decGi16   = (_Float16*)alloc((size_t)T * B * 3 * DDEC * 2);
  _Float16* mem16     = (_Float16*)alloc((size_t)S * B * DDEC * 2);
  _Float16* k16       = (_Float16*)alloc((size_t)S * B * DDEC * 2);
  _Float16* v16       = (_Float16*)alloc((size_t)S * B * DDEC * 2);
  _Float16* hs16      = (_Float16*)alloc((size_t)T * B * DDEC * 2);
  float*    ghbuf     = (float*)alloc((size_t)B * 3 * DDEC * 4);

  auto conv = [&](const float* s, _Float16* d, int n) {
    int blocks = (n + 255) / 256;
    if (blocks > 16384) blocks = 16384;
    k_f32_to_f16<<<blocks, 256, 0, stream>>>(s, d, n);
  };
  conv(enc_Wih_f, encWihF16, 768 * 128);
  conv(enc_Whh_f, encWhhF16, 768 * 256);
  conv(enc_Wih_b, encWihB16, 768 * 128);
  conv(enc_Whh_b, encWhhB16, 768 * 256);
  conv(dec_Wih,   decWih16,  1536 * 128);
  conv(dec_Whh,   decWhh16,  1536 * 512);
  conv(attn_in_w, attnIn16,  1536 * 512);
  conv(attn_out_w,attnOut16, 512 * 512);
  conv(proj_w,    projW16,   32000 * 512);

  k_gather_enc<<<2048, 256, 0, stream>>>(nx, tok_emb, emb16);
  k_gather_dec<<<2048, 256, 0, stream>>>(x, tok_emb, start_emb, decin16);

  k_gemm_f16<<<1536, 256, 0, stream>>>(emb16, DE, encWihF16, DE, enc_bih_f,
                                       giF16, 3 * DENC, S * B, 3 * DENC, DE);
  k_gemm_f16<<<1536, 256, 0, stream>>>(emb16, DE, encWihB16, DE, enc_bih_b,
                                       giB16, 3 * DENC, S * B, 3 * DENC, DE);
  k_gemm_f16<<<3072, 256, 0, stream>>>(decin16, DE, decWih16, DE, dec_bih,
                                       decGi16, 3 * DDEC, T * B, 3 * DDEC, DE);

  k_enc_scan<<<2, 256, 0, stream>>>(encWhhF16, encWhhB16, enc_bhh_f, enc_bhh_b,
                                    giF16, giB16, mem16);

  k_gemm_f16<<<1024, 256, 0, stream>>>(mem16, DDEC, attnIn16 + (size_t)512 * 512, DDEC,
                                       attn_in_b + 512, k16, DDEC, S * B, DDEC, DDEC);
  k_gemm_f16<<<1024, 256, 0, stream>>>(mem16, DDEC, attnIn16 + (size_t)1024 * 512, DDEC,
                                       attn_in_b + 1024, v16, DDEC, S * B, DDEC, DDEC);

  k_dec_scan<<<1, 512, 0, stream>>>(decWhh16, dec_bhh, decGi16,
                                    attnIn16, attn_in_b,
                                    attnOut16, attn_out_b,
                                    k16, v16, ln_g, ln_b,
                                    style_emb, label, ghbuf, hs16);

  // 134 GFLOP projection: 8000 blocks of 128x128 WMMA macro-tiles
  k_gemm_proj<<<32 * 250, 256, 0, stream>>>(hs16, projW16, proj_b, out);
}
